// TGNmodel_61907658604944
// MI455X (gfx1250) — compile-verified, matrix operand order checked by reading, not run on hardware
//
#include <hip/hip_runtime.h>
#include <hip/hip_bf16.h>

// ---------------------------------------------------------------------------
// TGN forward for MI455X (gfx1250, wave32, WMMA).
// All GEMM work goes through v_wmma_f32_16x16x32_bf16 (f32 accumulate).
// Two-layer MLPs are fused per kernel; the tanh intermediate is re-staged
// through a per-wave LDS tile to convert C-fragment layout -> A-fragment
// layout. segment_sum is a fused global_atomic_add_f32 scatter.
// ---------------------------------------------------------------------------

typedef __bf16 bf16_t;
typedef __attribute__((ext_vector_type(16))) __bf16 v16bf;
typedef __attribute__((ext_vector_type(8)))  __bf16 v8bf;
typedef __attribute__((ext_vector_type(8)))  float  v8f;

#define DEV __device__ __forceinline__

// fast tanh: 1 - 2/(exp(2x)+1) via v_exp_f32 + v_rcp_f32 (saturates to +/-1)
DEV float tanh_fast(float x) {
  const float e = __builtin_amdgcn_exp2f(x * 2.88539008177792681f); // 2*log2(e)
  const float r = __builtin_amdgcn_rcpf(e + 1.0f);
  return fmaf(-2.0f, r, 1.0f);
}

DEV v16bf cat16(v8bf lo, v8bf hi) {
  return __builtin_shufflevector(lo, hi, 0, 1, 2, 3, 4, 5, 6, 7,
                                         8, 9, 10, 11, 12, 13, 14, 15);
}

// A/B fragment for 16x16x32 bf16 WMMA, loaded from a row-major row:
// lane holds K = {k0 + half*8 + 0..7} and {k0 + 16 + half*8 + 0..7}.
DEV v16bf ldfrag_bf16(const bf16_t* p) {
  v8bf lo = *(const v8bf*)(p);
  v8bf hi = *(const v8bf*)(p + 16);
  return cat16(lo, hi);
}

// Same fragment but source is f32; convert (with per-row scale) to bf16.
DEV v16bf ldfrag_f32(const float* p, float s) {
  float4 a = *(const float4*)(p);
  float4 b = *(const float4*)(p + 4);
  float4 c = *(const float4*)(p + 16);
  float4 d = *(const float4*)(p + 20);
  v16bf r;
  r[0]  = (bf16_t)(a.x * s); r[1]  = (bf16_t)(a.y * s);
  r[2]  = (bf16_t)(a.z * s); r[3]  = (bf16_t)(a.w * s);
  r[4]  = (bf16_t)(b.x * s); r[5]  = (bf16_t)(b.y * s);
  r[6]  = (bf16_t)(b.z * s); r[7]  = (bf16_t)(b.w * s);
  r[8]  = (bf16_t)(c.x * s); r[9]  = (bf16_t)(c.y * s);
  r[10] = (bf16_t)(c.z * s); r[11] = (bf16_t)(c.w * s);
  r[12] = (bf16_t)(d.x * s); r[13] = (bf16_t)(d.y * s);
  r[14] = (bf16_t)(d.z * s); r[15] = (bf16_t)(d.w * s);
  return r;
}

DEV v8f wmma_bf16(v16bf a, v16bf b, v8f c) {
  return __builtin_amdgcn_wmma_f32_16x16x32_bf16(
      /*neg_a=*/false, a, /*neg_b=*/false, b,
      /*c_mod=*/(short)0, c, /*reuse_a=*/false, /*reuse_b=*/false);
}

// ---------------------------------------------------------------------------
// Fused 2-layer MLP:  out = tanh(A @ W0 + b0) @ W1 + b1
// A (nRows x K) is assembled from up to 4 f32 sources of 64 cols each;
// tail sources may be row-gathered (idx) and scaled (val).
// Output: plain f32 store, or atomic scatter-add out[oidx[r]] += oval[r]*y[r].
// Weights are pre-transposed bf16:  w0T[n][k] (n<64, k<K), w1T[n][k] (64x64).
// One wave = one 16-row x 64-col tile.  256 threads = 8 waves = 128 rows.
// ---------------------------------------------------------------------------
template <int KT, bool GATHER, bool SCATTER>
__global__ __launch_bounds__(256) void tgn_mlp2(
    const float* __restrict__ s0, const float* __restrict__ s1,
    const float* __restrict__ s2, const float* __restrict__ s3,
    const int* __restrict__ gi1, const int* __restrict__ gi2,
    const int* __restrict__ gi3,
    const float* __restrict__ gv1, const float* __restrict__ gv2,
    const float* __restrict__ gv3,
    const bf16_t* __restrict__ w0T, const float* __restrict__ b0,
    const bf16_t* __restrict__ w1T, const float* __restrict__ b1,
    float* __restrict__ out, const int* __restrict__ oidx,
    const float* __restrict__ oval, int nRows) {
  constexpr int K = KT * 64;
  __shared__ bf16_t hbuf[8][16][64];  // per-wave 16x64 bf16 tanh tile (16 KB)

  const int lane = threadIdx.x & 31;
  const int wv   = threadIdx.x >> 5;
  const int half = lane >> 4;
  const int l15  = lane & 15;
  const int m0   = (blockIdx.x * 8 + wv) * 16;
  if (m0 >= nRows) return;  // wave-uniform: EXEC stays all-ones for WMMA

  const int rAc = min(m0 + l15, nRows - 1);  // clamped A-row for this lane
  const float* rowp0 = s0 + (size_t)rAc * 64;
  const float* rowp1 = nullptr;
  const float* rowp2 = nullptr;
  const float* rowp3 = nullptr;
  float sc1 = 1.f, sc2 = 1.f, sc3 = 1.f;
  if constexpr (KT >= 2) {
    if constexpr (GATHER) {
      rowp1 = s1 + (size_t)gi1[rAc] * 64; sc1 = gv1[rAc];
    } else {
      rowp1 = s1 + (size_t)rAc * 64;
    }
  }
  if constexpr (KT >= 4) {
    if constexpr (GATHER) {
      rowp2 = s2 + (size_t)gi2[rAc] * 64; sc2 = gv2[rAc];
      rowp3 = s3 + (size_t)gi3[rAc] * 64; sc3 = gv3[rAc];
    } else {
      rowp2 = s2 + (size_t)rAc * 64;
      rowp3 = s3 + (size_t)rAc * 64;
    }
  }

  // ---- layer 1: acc = A @ W0 + b0 ----
  v8f acc[4];
#pragma unroll
  for (int t = 0; t < 4; ++t) {
    const float bb = b0[t * 16 + l15];
#pragma unroll
    for (int j = 0; j < 8; ++j) acc[t][j] = bb;
  }

#pragma unroll
  for (int kc = 0; kc < 2 * KT; ++kc) {
    const int k0 = kc * 32;
    const int si = k0 >> 6;                 // which 64-col source
    const int lk = (k0 & 63) + half * 8;    // lane offset within source row
    v16bf a;
    if (si == 0)      a = ldfrag_f32(rowp0 + lk, 1.0f);
    else if (si == 1) a = ldfrag_f32(rowp1 + lk, sc1);
    else if (si == 2) a = ldfrag_f32(rowp2 + lk, sc2);
    else              a = ldfrag_f32(rowp3 + lk, sc3);
#pragma unroll
    for (int t = 0; t < 4; ++t) {
      v16bf b = ldfrag_bf16(w0T + (size_t)(t * 16 + l15) * K + k0 + half * 8);
      acc[t] = wmma_bf16(a, b, acc[t]);
    }
  }

  // ---- tanh, C-fragment -> LDS row-major tile (bf16) ----
#pragma unroll
  for (int t = 0; t < 4; ++t)
#pragma unroll
    for (int j = 0; j < 8; ++j)
      hbuf[wv][j + 8 * half][t * 16 + l15] = (bf16_t)tanh_fast(acc[t][j]);

  // ---- layer 2: acc2 = H @ W1 + b1 (A-fragments re-read from LDS) ----
  v8f acc2[4];
#pragma unroll
  for (int t = 0; t < 4; ++t) {
    const float bb = b1[t * 16 + l15];
#pragma unroll
    for (int j = 0; j < 8; ++j) acc2[t][j] = bb;
  }
#pragma unroll
  for (int kc = 0; kc < 2; ++kc) {
    const int k0 = kc * 32;
    v16bf a = ldfrag_bf16(&hbuf[wv][l15][0] + k0 + half * 8);
#pragma unroll
    for (int t = 0; t < 4; ++t) {
      v16bf b = ldfrag_bf16(w1T + (size_t)(t * 16 + l15) * 64 + k0 + half * 8);
      acc2[t] = wmma_bf16(a, b, acc2[t]);
    }
  }

  // ---- output ----
  if constexpr (SCATTER) {
    int sgi[8]; float sgv[8];
#pragma unroll
    for (int j = 0; j < 8; ++j) {
      const int rc = min(m0 + j + 8 * half, nRows - 1);
      sgi[j] = oidx[rc];
      sgv[j] = oval[rc];
    }
#pragma unroll
    for (int t = 0; t < 4; ++t)
#pragma unroll
      for (int j = 0; j < 8; ++j) {
        const int r = m0 + j + 8 * half;
        if (r < nRows)
          unsafeAtomicAdd(out + (size_t)sgi[j] * 64 + t * 16 + l15,
                          sgv[j] * acc2[t][j]);
      }
  } else {
#pragma unroll
    for (int t = 0; t < 4; ++t)
#pragma unroll
      for (int j = 0; j < 8; ++j) {
        const int r = m0 + j + 8 * half;
        if (r < nRows) out[(size_t)r * 64 + t * 16 + l15] = acc2[t][j];
      }
  }
}

// ---------------------------------------------------------------------------
// Small helper kernels
// ---------------------------------------------------------------------------
__global__ void tgn_zero(float* __restrict__ p, int n) {
  int i = blockIdx.x * blockDim.x + threadIdx.x;
  if (i < n) p[i] = 0.f;
}

// encoder: out[r][c] = b[c] + sum_{k<4} x[r][k]*w[k][c]
__global__ void tgn_encode(const float* __restrict__ x,
                           const float* __restrict__ w,
                           const float* __restrict__ b,
                           float* __restrict__ out, int n) {
  int i = blockIdx.x * blockDim.x + threadIdx.x;
  if (i >= n * 64) return;
  int r = i >> 6, c = i & 63;
  const float4 xr = *(const float4*)(x + (size_t)r * 4);
  float acc = b[c];
  acc = fmaf(xr.x, w[0 * 64 + c], acc);
  acc = fmaf(xr.y, w[1 * 64 + c], acc);
  acc = fmaf(xr.z, w[2 * 64 + c], acc);
  acc = fmaf(xr.w, w[3 * 64 + c], acc);
  out[i] = acc;
}

// weight prep: wT[n][k] = (bf16) w[k][n]   (w is K x 64 row-major)
__global__ void tgn_wprep(const float* __restrict__ w, bf16_t* __restrict__ wT,
                          int K) {
  int i = blockIdx.x * blockDim.x + threadIdx.x;
  if (i >= 64 * K) return;
  int n = i / K, k = i - n * K;
  wT[i] = (bf16_t)w[(size_t)k * 64 + n];
}

// decoder: out[r][o] = db[o] + sum_c s[r][c]*dw[c][o], o < 2
__global__ void tgn_decode(const float* __restrict__ s,
                           const float* __restrict__ dw,
                           const float* __restrict__ db,
                           float* __restrict__ out, int n) {
  int i = blockIdx.x * blockDim.x + threadIdx.x;
  if (i >= n * 2) return;
  int r = i >> 1, o = i & 1;
  const float* sr = s + (size_t)r * 64;
  float acc = db[o];
#pragma unroll
  for (int c4 = 0; c4 < 16; ++c4) {
    const float4 v = *(const float4*)(sr + c4 * 4);
    acc = fmaf(v.x, dw[(c4 * 4 + 0) * 2 + o], acc);
    acc = fmaf(v.y, dw[(c4 * 4 + 1) * 2 + o], acc);
    acc = fmaf(v.z, dw[(c4 * 4 + 2) * 2 + o], acc);
    acc = fmaf(v.w, dw[(c4 * 4 + 3) * 2 + o], acc);
  }
  out[i] = acc;
}

// ---------------------------------------------------------------------------
// Host launcher
// ---------------------------------------------------------------------------
extern "C" void kernel_launch(void* const* d_in, const int* in_sizes, int n_in,
                              void* d_out, int out_size, void* d_ws,
                              size_t ws_size, hipStream_t stream) {
  (void)in_sizes; (void)out_size; (void)ws_size;
  if (n_in < 54) return;

  constexpr int N_PV = 25000, N_PQ = 100000, N_S = 500, N_E = 250000, T = 5;

  // ---- inputs (setup_inputs dict order, params flattened in insert order) --
  const float* x_pv = (const float*)d_in[0];
  const float* x_pq = (const float*)d_in[1];
  const float* x_s  = (const float*)d_in[2];
  const float* x_e  = (const float*)d_in[3];
  const float* val_pv = (const float*)d_in[4];
  const float* val_pq = (const float*)d_in[5];
  const float* val_s  = (const float*)d_in[6];
  const int* row_pv = (const int*)d_in[7];
  const int* row_pq = (const int*)d_in[8];
  const int* row_s  = (const int*)d_in[9];
  const int P = 10;                       // enc_w_pv, enc_b_pv, ... (8)
  const int PM = P + 8;                   // 5 msg MLPs x (w0,b0,w1,b1)
  const int PU = PM + 20;                 // 3 upd MLPs x (w0,b0,w1,b1)
  const int PD = PU + 12;                 // dec_w_pv, dec_b_pv, dec_w_pq, dec_b_pq
  auto F = [&](int i) { return (const float*)d_in[i]; };

  // ---- workspace carve-up ----
  char* ws = (char*)d_ws;
  size_t off = 0;
  auto galloc = [&](size_t bytes) -> void* {
    void* p = ws + off;
    off = (off + bytes + 255) & ~(size_t)255;
    return p;
  };
  float* s_pv = (float*)galloc((size_t)N_PV * 64 * 4);
  float* s_pq = (float*)galloc((size_t)N_PQ * 64 * 4);
  float* s_s  = (float*)galloc((size_t)N_S  * 64 * 4);
  float* s_e  = (float*)galloc((size_t)N_E  * 64 * 4);
  float* m_pv = (float*)galloc((size_t)N_PV * 64 * 4);
  float* m_pq = (float*)galloc((size_t)N_PQ * 64 * 4);
  float* z_pv = (float*)galloc((size_t)N_PV * 64 * 4);
  float* z_pq = (float*)galloc((size_t)N_PQ * 64 * 4);
  float* z_s  = (float*)galloc((size_t)N_S  * 64 * 4);

  bf16_t* msg0T[5];
  bf16_t* msg1T[5];
  for (int i = 0; i < 5; ++i) {
    msg0T[i] = (bf16_t*)galloc(64 * 64 * 2);
    msg1T[i] = (bf16_t*)galloc(64 * 64 * 2);
  }
  bf16_t* updpv0T = (bf16_t*)galloc(64 * 128 * 2);
  bf16_t* updpv1T = (bf16_t*)galloc(64 * 64 * 2);
  bf16_t* updpq0T = (bf16_t*)galloc(64 * 128 * 2);
  bf16_t* updpq1T = (bf16_t*)galloc(64 * 64 * 2);
  bf16_t* upde0T  = (bf16_t*)galloc(64 * 256 * 2);
  bf16_t* upde1T  = (bf16_t*)galloc(64 * 64 * 2);

  auto cdiv = [](int a, int b) { return (a + b - 1) / b; };

  // ---- weight conversion/transpose to bf16 (every call; deterministic) ----
  struct WP { const float* w; bf16_t* wT; int K; };
  WP wp[16] = {
      {F(PM + 0),  msg0T[0], 64}, {F(PM + 2),  msg1T[0], 64},   // e2pv
      {F(PM + 4),  msg0T[1], 64}, {F(PM + 6),  msg1T[1], 64},   // e2pq
      {F(PM + 8),  msg0T[2], 64}, {F(PM + 10), msg1T[2], 64},   // pv2e
      {F(PM + 12), msg0T[3], 64}, {F(PM + 14), msg1T[3], 64},   // pq2e
      {F(PM + 16), msg0T[4], 64}, {F(PM + 18), msg1T[4], 64},   // s2e
      {F(PU + 0),  updpv0T, 128}, {F(PU + 2),  updpv1T, 64},
      {F(PU + 4),  updpq0T, 128}, {F(PU + 6),  updpq1T, 64},
      {F(PU + 8),  upde0T,  256}, {F(PU + 10), upde1T,  64},
  };
  for (int i = 0; i < 16; ++i)
    tgn_wprep<<<cdiv(64 * wp[i].K, 256), 256, 0, stream>>>(wp[i].w, wp[i].wT,
                                                           wp[i].K);

  // ---- encoders ----
  tgn_encode<<<cdiv(N_PV * 64, 256), 256, 0, stream>>>(x_pv, F(P + 0), F(P + 1), s_pv, N_PV);
  tgn_encode<<<cdiv(N_PQ * 64, 256), 256, 0, stream>>>(x_pq, F(P + 2), F(P + 3), s_pq, N_PQ);
  tgn_encode<<<cdiv(N_S  * 64, 256), 256, 0, stream>>>(x_s,  F(P + 4), F(P + 5), s_s,  N_S);
  tgn_encode<<<cdiv(N_E  * 64, 256), 256, 0, stream>>>(x_e,  F(P + 6), F(P + 7), s_e,  N_E);

  // z_s = mlp2(s_s, s2e): s_s never changes -> hoisted out of the time loop
  tgn_mlp2<1, false, false><<<cdiv(N_S, 128), 256, 0, stream>>>(
      s_s, nullptr, nullptr, nullptr, nullptr, nullptr, nullptr, nullptr,
      nullptr, nullptr, msg0T[4], F(PM + 17), msg1T[4], F(PM + 19), z_s,
      nullptr, nullptr, N_S);

  // ---- T message-passing steps ----
  for (int t = 0; t < T; ++t) {
    tgn_zero<<<cdiv(N_PV * 64, 256), 256, 0, stream>>>(m_pv, N_PV * 64);
    tgn_zero<<<cdiv(N_PQ * 64, 256), 256, 0, stream>>>(m_pq, N_PQ * 64);

    // m_pv += scatter(val_pv * mlp2(s_e, e2pv), row_pv)
    tgn_mlp2<1, false, true><<<cdiv(N_E, 128), 256, 0, stream>>>(
        s_e, nullptr, nullptr, nullptr, nullptr, nullptr, nullptr, nullptr,
        nullptr, nullptr, msg0T[0], F(PM + 1), msg1T[0], F(PM + 3), m_pv,
        row_pv, val_pv, N_E);
    tgn_mlp2<1, false, true><<<cdiv(N_E, 128), 256, 0, stream>>>(
        s_e, nullptr, nullptr, nullptr, nullptr, nullptr, nullptr, nullptr,
        nullptr, nullptr, msg0T[1], F(PM + 5), msg1T[1], F(PM + 7), m_pq,
        row_pq, val_pq, N_E);

    // z_pv = mlp2(s_pv, pv2e), z_pq = mlp2(s_pq, pq2e)  (old states)
    tgn_mlp2<1, false, false><<<cdiv(N_PV, 128), 256, 0, stream>>>(
        s_pv, nullptr, nullptr, nullptr, nullptr, nullptr, nullptr, nullptr,
        nullptr, nullptr, msg0T[2], F(PM + 9), msg1T[2], F(PM + 11), z_pv,
        nullptr, nullptr, N_PV);
    tgn_mlp2<1, false, false><<<cdiv(N_PQ, 128), 256, 0, stream>>>(
        s_pq, nullptr, nullptr, nullptr, nullptr, nullptr, nullptr, nullptr,
        nullptr, nullptr, msg0T[3], F(PM + 13), msg1T[3], F(PM + 15), z_pq,
        nullptr, nullptr, N_PQ);

    // node updates (in-place; z_* already consumed old states)
    tgn_mlp2<2, false, false><<<cdiv(N_PV, 128), 256, 0, stream>>>(
        s_pv, m_pv, nullptr, nullptr, nullptr, nullptr, nullptr, nullptr,
        nullptr, nullptr, updpv0T, F(PU + 1), updpv1T, F(PU + 3), s_pv,
        nullptr, nullptr, N_PV);
    tgn_mlp2<2, false, false><<<cdiv(N_PQ, 128), 256, 0, stream>>>(
        s_pq, m_pq, nullptr, nullptr, nullptr, nullptr, nullptr, nullptr,
        nullptr, nullptr, updpq0T, F(PU + 5), updpq1T, F(PU + 7), s_pq,
        nullptr, nullptr, N_PQ);

    // edge update: A = [s_e | val_pv*z_pv[row_pv] | val_pq*z_pq[row_pq] | val_s*z_s[row_s]]
    tgn_mlp2<4, true, false><<<cdiv(N_E, 128), 256, 0, stream>>>(
        s_e, z_pv, z_pq, z_s, row_pv, row_pq, row_s, val_pv, val_pq, val_s,
        upde0T, F(PU + 9), upde1T, F(PU + 11), s_e, nullptr, nullptr, N_E);
  }

  // ---- decoders: d_out = [out_pv (25000x2) | out_pq (100000x2)] ----
  float* out = (float*)d_out;
  tgn_decode<<<cdiv(N_PV * 2, 256), 256, 0, stream>>>(s_pv, F(PD + 0),
                                                      F(PD + 1), out, N_PV);
  tgn_decode<<<cdiv(N_PQ * 2, 256), 256, 0, stream>>>(
      s_pq, F(PD + 2), F(PD + 3), out + (size_t)N_PV * 2, N_PQ);
}
